// MultiHeadAttention_45157286150187
// MI455X (gfx1250) — compile-verified, matrix operand order checked by reading, not run on hardware
//
#include <hip/hip_runtime.h>
#include <hip/hip_bf16.h>

// ---------------------------------------------------------------------------
// Types for CDNA5 WMMA (wave32): v_wmma_f32_16x16x32_bf16
// ---------------------------------------------------------------------------
typedef __attribute__((ext_vector_type(16))) __bf16 v16bf;
typedef __attribute__((ext_vector_type(8)))  float  v8f;

__device__ __forceinline__ unsigned short f32_to_bf16(float f) {
    unsigned int u = __float_as_uint(f);
    unsigned int r = (u + 0x7fffu + ((u >> 16) & 1u)) >> 16;
    return (unsigned short)r;
}

// Low 32 bits of a flat shared-memory address = LDS byte offset (LDS aperture).
__device__ __forceinline__ unsigned lds_off(const void* p) {
    return (unsigned)(unsigned long long)p;
}

// CDNA5 async copy: global -> LDS, 16B per lane, tracked by ASYNCcnt.
__device__ __forceinline__ void async_b128(unsigned loff, const unsigned short* g) {
    asm volatile("global_load_async_to_lds_b128 %0, %1, off"
                 :: "v"(loff), "v"((unsigned long long)g)
                 : "memory");
}

// Load one 16x32 bf16 fragment in the CDNA5 A-operand layout from a row-major
// matrix (also used for the B operand with transposed [N][K] storage).
// Lane L (0-15): row L, K = {0..7, 16..23}; lane L+16: row L, K = {8..15, 24..31}.
__device__ __forceinline__ v16bf load_frag(const unsigned short* base, int ld, int lane) {
    const int row = lane & 15;
    const int k0  = (lane & 16) ? 8 : 0;
    const unsigned short* p = base + (size_t)row * ld + k0;
    union { v16bf v; uint4 u[2]; } f;
    f.u[0] = *(const uint4*)(p);
    f.u[1] = *(const uint4*)(p + 16);
    return f.v;
}

__device__ __forceinline__ v8f wmma_bf16(v16bf a, v16bf b, v8f c) {
    return __builtin_amdgcn_wmma_f32_16x16x32_bf16(false, a, false, b, (short)0, c, false, false);
}

// ---------------------------------------------------------------------------
// Elementwise f32 -> bf16 conversion
// ---------------------------------------------------------------------------
__global__ __launch_bounds__(256) void cvt_bf16_kernel(const float* __restrict__ in,
                                                       unsigned short* __restrict__ out,
                                                       size_t n) {
    size_t i = (size_t)blockIdx.x * blockDim.x + threadIdx.x;
    if (i < n) out[i] = f32_to_bf16(in[i]);
}

// w [K][N] f32  ->  wT [N][K] bf16   (32x32 LDS tiles)
__global__ __launch_bounds__(256) void transpose_cvt_kernel(const float* __restrict__ w,
                                                            unsigned short* __restrict__ wT,
                                                            int Kdim, int Ndim) {
    __shared__ unsigned short t[32][33];
    const int k0 = blockIdx.y * 32, n0 = blockIdx.x * 32;
    const int tx = threadIdx.x & 31, ty = threadIdx.x >> 5;
#pragma unroll
    for (int i = 0; i < 32; i += 8)
        t[ty + i][tx] = f32_to_bf16(w[(size_t)(k0 + ty + i) * Ndim + n0 + tx]);
    __syncthreads();
#pragma unroll
    for (int i = 0; i < 32; i += 8)
        wT[(size_t)(n0 + ty + i) * Kdim + k0 + tx] = t[tx][ty + i];
}

// V heads [BH][S][64] bf16 -> Vt [BH][64][S] bf16
__global__ __launch_bounds__(256) void transpose_v_kernel(const unsigned short* __restrict__ v,
                                                          unsigned short* __restrict__ vt,
                                                          int S) {
    __shared__ unsigned short t[32][33];
    const int head = blockIdx.z;
    const int s0 = blockIdx.x * 32, d0 = blockIdx.y * 32;
    const size_t hb = (size_t)head * S * 64;
    const int tx = threadIdx.x & 31, ty = threadIdx.x >> 5;
#pragma unroll
    for (int i = 0; i < 32; i += 8)
        t[ty + i][tx] = v[hb + (size_t)(s0 + ty + i) * 64 + d0 + tx];
    __syncthreads();
#pragma unroll
    for (int i = 0; i < 32; i += 8)
        vt[hb + (size_t)(d0 + ty + i) * S + s0 + tx] = t[tx][ty + i];
}

// ---------------------------------------------------------------------------
// Async double-buffered bf16 GEMM: C[M][N] = A[M][K] @ BT[N][K]^T (+bias)(+res)
// WG = 256 threads (8 waves), tile 128x128, per-wave 64x32, K-step 32.
// Tiles stream global->LDS with GLOBAL_LOAD_ASYNC_TO_LDS_B128 (ASYNCcnt).
// mode 0: bf16 output;  mode 1: f32 output with residual add.
// ---------------------------------------------------------------------------
#define GM_BM 128
#define GM_BN 128
#define GM_BK 32
#define GM_LDA 40                                    // padded LDS stride (ushorts)
#define GM_TILE_ELEMS ((GM_BM + GM_BN) * GM_LDA)     // one buffer (A then B)

__device__ __forceinline__ void gemm_issue(const unsigned short* A, const unsigned short* BT,
                                           int K, int m0, int n0, int kk,
                                           unsigned base, int tid) {
#pragma unroll
    for (int j = 0; j < 4; ++j) {                    // 4 async instrs / thread
        const int idx = tid + j * 256;               // 0..1023
        if (idx < GM_BM * 4) {                       // A tile: 128 rows x 4 chunks
            const int r = idx >> 2, c = (idx & 3) * 8;
            async_b128(base + (unsigned)((r * GM_LDA + c) * 2),
                       A + (size_t)(m0 + r) * K + kk + c);
        } else {                                     // B tile: 128 rows x 4 chunks
            const int t = idx - GM_BM * 4;
            const int r = t >> 2, c = (t & 3) * 8;
            async_b128(base + (unsigned)(((GM_BM + r) * GM_LDA + c) * 2),
                       BT + (size_t)(n0 + r) * K + kk + c);
        }
    }
}

__global__ __launch_bounds__(256) void gemm_bf16_kernel(
    const unsigned short* __restrict__ A,
    const unsigned short* __restrict__ BT,
    const float* __restrict__ bias,
    const float* __restrict__ residual,
    unsigned short* __restrict__ Cb,
    float* __restrict__ Cf,
    int M, int N, int K, int mode) {
    __shared__ unsigned short smem[2 * GM_TILE_ELEMS];

    const int tid  = threadIdx.x;
    const int lane = tid & 31;
    const int wave = tid >> 5;
    const int m0 = blockIdx.y * GM_BM;
    const int n0 = blockIdx.x * GM_BN;
    const int wm = (wave >> 2) * 64;   // 2 waves along M
    const int wn = (wave & 3) * 32;    // 4 waves along N

    const unsigned smem0 = lds_off(&smem[0]);

    v8f acc[4][2] = {};

    gemm_issue(A, BT, K, m0, n0, 0, smem0, tid);     // prologue: tile 0 -> buf 0
    int cur = 0;
    for (int kk = 0; kk < K; kk += GM_BK) {
        if (kk + GM_BK < K) {
            gemm_issue(A, BT, K, m0, n0, kk + GM_BK,
                       smem0 + (unsigned)((cur ^ 1) * GM_TILE_ELEMS * 2), tid);
            asm volatile("s_wait_asynccnt 0x4" ::: "memory");   // current group done
        } else {
            asm volatile("s_wait_asynccnt 0x0" ::: "memory");
        }
        __syncthreads();   // all waves' chunks of current buffer resident

        const unsigned short* Abase = &smem[cur * GM_TILE_ELEMS];
        const unsigned short* Bbase = Abase + GM_BM * GM_LDA;
        v16bf af[4], bfr[2];
#pragma unroll
        for (int i = 0; i < 4; ++i)
            af[i] = load_frag(Abase + (wm + i * 16) * GM_LDA, GM_LDA, lane);
#pragma unroll
        for (int j = 0; j < 2; ++j)
            bfr[j] = load_frag(Bbase + (wn + j * 16) * GM_LDA, GM_LDA, lane);
#pragma unroll
        for (int i = 0; i < 4; ++i)
#pragma unroll
            for (int j = 0; j < 2; ++j)
                acc[i][j] = wmma_bf16(af[i], bfr[j], acc[i][j]);

        __syncthreads();   // reads done before other buffer gets overwritten
        cur ^= 1;
    }

    // Epilogue. C layout: (m,n): VGPR r, m = r + 8*(lane>=16), n = lane&15.
    const int col   = lane & 15;
    const int rbase = (lane >> 4) << 3;
#pragma unroll
    for (int i = 0; i < 4; ++i) {
#pragma unroll
        for (int j = 0; j < 2; ++j) {
            const int n = n0 + wn + j * 16 + col;
            const float bv = bias ? bias[n] : 0.0f;
#pragma unroll
            for (int r = 0; r < 8; ++r) {
                const int m = m0 + wm + i * 16 + rbase + r;
                float v = acc[i][j][r] + bv;
                if (mode == 0) {
                    Cb[(size_t)m * N + n] = f32_to_bf16(v);
                } else {
                    Cf[(size_t)m * N + n] = v + residual[(size_t)m * N + n];
                }
            }
        }
    }
}

// ---------------------------------------------------------------------------
// Flash-style attention with async K/V staging through LDS.
// WG = 128 threads (4 waves) = one (head, 64-query block); 16 rows per wave.
// Q,K: bf16 [BH][S][64]; Vt: bf16 [BH][64][S]; Ctx: bf16 [BH][S][64].
// Per key-block (32 keys): K tile 32x64 and Vt tile 64x32 staged via ASYNCcnt
// double-buffering and shared by all 4 waves.
// ---------------------------------------------------------------------------
#define AT_KT_LD 72
#define AT_VT_LD 40
#define AT_BUF_ELEMS (32 * AT_KT_LD + 64 * AT_VT_LD)   // 4864 ushorts / buffer
#define AT_PS_OFF (2 * AT_BUF_ELEMS)                   // P staging after buffers

__device__ __forceinline__ void attn_issue(const unsigned short* Kh, const unsigned short* Vth,
                                           int S, int kb, unsigned base, int tid) {
#pragma unroll
    for (int j = 0; j < 4; ++j) {                      // 4 async instrs / thread
        const int idx = tid + j * 128;                 // 0..511
        if (idx < 256) {                               // K tile: 32 rows x 8 chunks
            const int r = idx >> 3, c = (idx & 7) * 8;
            async_b128(base + (unsigned)((r * AT_KT_LD + c) * 2),
                       Kh + (size_t)(kb + r) * 64 + c);
        } else {                                       // Vt tile: 64 rows x 4 chunks
            const int t = idx - 256;
            const int r = t >> 2, c = (t & 3) * 8;
            async_b128(base + (unsigned)((32 * AT_KT_LD + r * AT_VT_LD + c) * 2),
                       Vth + (size_t)r * S + kb + c);
        }
    }
}

__global__ __launch_bounds__(128) void attn_kernel(
    const unsigned short* __restrict__ Qb,
    const unsigned short* __restrict__ Kb,
    const unsigned short* __restrict__ Vt,
    unsigned short* __restrict__ Ctx,
    int S) {
    __shared__ unsigned short asmem[2 * AT_BUF_ELEMS + 4 * 16 * 40];

    const int head = blockIdx.y;
    const int tid  = threadIdx.x;
    const int lane = tid & 31;
    const int wave = tid >> 5;
    const size_t hq = (size_t)head * S * 64;
    const size_t hv = (size_t)head * 64 * S;
    const int q0 = blockIdx.x * 64 + wave * 16;

    const unsigned short* Kh  = Kb + hq;
    const unsigned short* Vth = Vt + hv;
    const unsigned base0 = lds_off(&asmem[0]);

    // Query fragments for this wave's 16 rows (D = 64 = two K=32 chunks).
    const v16bf qf0 = load_frag(Qb + hq + (size_t)q0 * 64,      64, lane);
    const v16bf qf1 = load_frag(Qb + hq + (size_t)q0 * 64 + 32, 64, lane);

    v8f acc[4] = {};
    float mrun[8], lrun[8];
#pragma unroll
    for (int r = 0; r < 8; ++r) { mrun[r] = -3.0e38f; lrun[r] = 0.0f; }

    const int   col     = lane & 15;
    const int   halfsel = lane >> 4;
    const float scale   = 0.125f;    // 1/sqrt(64)

    attn_issue(Kh, Vth, S, 0, base0, tid);             // prologue: block 0 -> buf 0
    int cur = 0;
    for (int kb = 0; kb < S; kb += 32) {
        if (kb + 32 < S) {
            attn_issue(Kh, Vth, S, kb + 32,
                       base0 + (unsigned)((cur ^ 1) * AT_BUF_ELEMS * 2), tid);
            asm volatile("s_wait_asynccnt 0x4" ::: "memory");
        } else {
            asm volatile("s_wait_asynccnt 0x0" ::: "memory");
        }
        __syncthreads();

        const unsigned short* Kt  = &asmem[cur * AT_BUF_ELEMS];
        const unsigned short* Vtt = Kt + 32 * AT_KT_LD;

        // scores: Q(16x64) @ K^T(64x32) -> two 16x16 f32 tiles
        v8f s0 = {}, s1 = {};
        {
            const v16bf k00 = load_frag(Kt,                     AT_KT_LD, lane);
            const v16bf k01 = load_frag(Kt + 32,                AT_KT_LD, lane);
            const v16bf k10 = load_frag(Kt + 16 * AT_KT_LD,      AT_KT_LD, lane);
            const v16bf k11 = load_frag(Kt + 16 * AT_KT_LD + 32, AT_KT_LD, lane);
            s0 = wmma_bf16(qf0, k00, s0);
            s0 = wmma_bf16(qf1, k01, s0);
            s1 = wmma_bf16(qf0, k10, s1);
            s1 = wmma_bf16(qf1, k11, s1);
        }

        // Online softmax per row (row = r + 8*halfsel; reduce across 16 lanes).
        unsigned short* Pw = &asmem[AT_PS_OFF + wave * 16 * 40];
#pragma unroll
        for (int r = 0; r < 8; ++r) {
            float a = s0[r] * scale;
            float b = s1[r] * scale;
            float mx = fmaxf(a, b);
#pragma unroll
            for (int off = 1; off < 16; off <<= 1) mx = fmaxf(mx, __shfl_xor(mx, off));
            const float mnew  = fmaxf(mrun[r], mx);
            const float alpha = __expf(mrun[r] - mnew);
            const float p0 = __expf(a - mnew);
            const float p1 = __expf(b - mnew);
            float ps = p0 + p1;
#pragma unroll
            for (int off = 1; off < 16; off <<= 1) ps += __shfl_xor(ps, off);
            lrun[r] = lrun[r] * alpha + ps;
            mrun[r] = mnew;
#pragma unroll
            for (int nf = 0; nf < 4; ++nf) acc[nf][r] *= alpha;
            Pw[(r + halfsel * 8) * 40 + col]      = f32_to_bf16(p0);
            Pw[(r + halfsel * 8) * 40 + col + 16] = f32_to_bf16(p1);
        }

        // PV: P(16x32) @ V(32x64). Per-wave LDS ops stay in order.
        const v16bf pf = load_frag(Pw, 40, lane);
#pragma unroll
        for (int nf = 0; nf < 4; ++nf) {
            const v16bf vf = load_frag(Vtt + nf * 16 * AT_VT_LD, AT_VT_LD, lane);
            acc[nf] = wmma_bf16(pf, vf, acc[nf]);
        }

        __syncthreads();   // reads done before other buffer is overwritten
        cur ^= 1;
    }

    // Normalize and store ctx (bf16).
#pragma unroll
    for (int r = 0; r < 8; ++r) {
        const float inv = 1.0f / lrun[r];
        const int m = q0 + r + halfsel * 8;
#pragma unroll
        for (int nf = 0; nf < 4; ++nf) {
            Ctx[hq + (size_t)m * 64 + nf * 16 + col] = f32_to_bf16(acc[nf][r] * inv);
        }
    }
}

// ---------------------------------------------------------------------------
// LayerNorm over E=1024, one 256-thread WG per row.
// ---------------------------------------------------------------------------
__global__ __launch_bounds__(256) void ln_kernel(const float* __restrict__ y,
                                                 const float* __restrict__ gamma,
                                                 const float* __restrict__ beta,
                                                 float* __restrict__ out, int E) {
    const int row = blockIdx.x;
    const float* yr = y + (size_t)row * E;
    float s = 0.0f, s2 = 0.0f;
    for (int i = threadIdx.x; i < E; i += blockDim.x) {
        float v = yr[i];
        s += v; s2 += v * v;
    }
#pragma unroll
    for (int off = 1; off < 32; off <<= 1) {
        s  += __shfl_xor(s,  off);
        s2 += __shfl_xor(s2, off);
    }
    __shared__ float red[2][8];
    const int wave = threadIdx.x >> 5, lane = threadIdx.x & 31;
    if (lane == 0) { red[0][wave] = s; red[1][wave] = s2; }
    __syncthreads();
    if (wave == 0) {
        s  = (lane < 8) ? red[0][lane] : 0.0f;
        s2 = (lane < 8) ? red[1][lane] : 0.0f;
#pragma unroll
        for (int off = 1; off < 8; off <<= 1) {
            s  += __shfl_xor(s,  off);
            s2 += __shfl_xor(s2, off);
        }
        if (lane == 0) { red[0][0] = s; red[1][0] = s2; }
    }
    __syncthreads();
    const float mean = red[0][0] / (float)E;
    const float var  = red[1][0] / (float)E - mean * mean;
    const float inv  = rsqrtf(var + 1e-5f);
    for (int i = threadIdx.x; i < E; i += blockDim.x)
        out[(size_t)row * E + i] = (yr[i] - mean) * inv * gamma[i] + beta[i];
}

// ---------------------------------------------------------------------------
// Host launch
// ---------------------------------------------------------------------------
extern "C" void kernel_launch(void* const* d_in, const int* in_sizes, int n_in,
                              void* d_out, int out_size, void* d_ws, size_t ws_size,
                              hipStream_t stream) {
    (void)in_sizes; (void)n_in; (void)out_size; (void)ws_size;
    const float* x     = (const float*)d_in[0];
    const float* wq    = (const float*)d_in[1];
    const float* bq    = (const float*)d_in[2];
    const float* wk    = (const float*)d_in[3];
    const float* bk    = (const float*)d_in[4];
    const float* wv    = (const float*)d_in[5];
    const float* bv    = (const float*)d_in[6];
    const float* wo    = (const float*)d_in[7];
    const float* bo    = (const float*)d_in[8];
    const float* gamma = (const float*)d_in[9];
    const float* beta  = (const float*)d_in[10];

    const int S = 2048, E = 1024;
    const int M = 2 * S;        // 4096 rows
    const int BH = 32;          // B*H "heads" under the raw reshape

    char* wsb = (char*)d_ws;
    size_t off = 0;
    auto alloc = [&](size_t bytes) -> void* {
        void* p = wsb + off;
        off += (bytes + 255) & ~(size_t)255;
        return p;
    };
    unsigned short* xb  = (unsigned short*)alloc((size_t)M * E * 2);
    unsigned short* wqT = (unsigned short*)alloc((size_t)E * E * 2);
    unsigned short* wkT = (unsigned short*)alloc((size_t)E * E * 2);
    unsigned short* wvT = (unsigned short*)alloc((size_t)E * E * 2);
    unsigned short* woT = (unsigned short*)alloc((size_t)E * E * 2);
    unsigned short* Qb  = (unsigned short*)alloc((size_t)M * E * 2);
    unsigned short* Kb  = (unsigned short*)alloc((size_t)M * E * 2);
    unsigned short* Vb  = (unsigned short*)alloc((size_t)M * E * 2);
    unsigned short* Vtb = (unsigned short*)alloc((size_t)M * E * 2);
    unsigned short* Ctx = (unsigned short*)alloc((size_t)M * E * 2);
    float*          y   = (float*)alloc((size_t)M * E * 4);

    {   // 1) x -> bf16
        size_t n = (size_t)M * E;
        cvt_bf16_kernel<<<dim3((unsigned)((n + 255) / 256)), dim3(256), 0, stream>>>(x, xb, n);
    }
    {   // 2) weights -> transposed bf16 [N][K]
        dim3 g(E / 32, E / 32), b(256);
        transpose_cvt_kernel<<<g, b, 0, stream>>>(wq, wqT, E, E);
        transpose_cvt_kernel<<<g, b, 0, stream>>>(wk, wkT, E, E);
        transpose_cvt_kernel<<<g, b, 0, stream>>>(wv, wvT, E, E);
        transpose_cvt_kernel<<<g, b, 0, stream>>>(wo, woT, E, E);
    }
    {   // 3) Q/K/V projections (bf16 out)
        dim3 g(E / GM_BN, M / GM_BM), b(256);
        gemm_bf16_kernel<<<g, b, 0, stream>>>(xb, wqT, bq, nullptr, Qb, nullptr, M, E, E, 0);
        gemm_bf16_kernel<<<g, b, 0, stream>>>(xb, wkT, bk, nullptr, Kb, nullptr, M, E, E, 0);
        gemm_bf16_kernel<<<g, b, 0, stream>>>(xb, wvT, bv, nullptr, Vb, nullptr, M, E, E, 0);
    }
    {   // 4) V head-transpose: [BH][S][64] -> [BH][64][S]
        dim3 g(S / 32, 64 / 32, BH), b(256);
        transpose_v_kernel<<<g, b, 0, stream>>>(Vb, Vtb, S);
    }
    {   // 5) attention
        dim3 g(S / 64, BH), b(128);
        attn_kernel<<<g, b, 0, stream>>>(Qb, Kb, Vtb, Ctx, S);
    }
    {   // 6) output projection + residual (f32)
        dim3 g(E / GM_BN, M / GM_BM), b(256);
        gemm_bf16_kernel<<<g, b, 0, stream>>>(Ctx, woT, bo, x, nullptr, y, M, E, E, 1);
    }
    // 7) LayerNorm
    ln_kernel<<<dim3(M), dim3(256), 0, stream>>>(y, gamma, beta, (float*)d_out, E);
}
